// ReModel_89395449299852
// MI455X (gfx1250) — matrix operand order repaired and checked by reading, not compiled, not run on hardware
//
#include <hip/hip_runtime.h>
#include <hip/hip_bf16.h>
#include <math.h>

#define Bb     4
#define LL     1024
#define HH     768
#define NHEADS 12
#define MM     128
#define EE     64
#define PP     2048
#define RR     97
#define BSs    64
#define GG     12
#define MAXV   65500.0f
#define MINV   6.103515625e-05f   // 2^-14
#define BPALL  (Bb * PP)          // 8192 pairs
#define RPAD   112                // R padded to 7 tiles of 16
#define WSTRIDE 72                // padded LDS row stride (bf16) for W slice

typedef __attribute__((ext_vector_type(16))) __bf16 v16bf;
typedef __attribute__((ext_vector_type(8)))  float  v8f;

__device__ __forceinline__ v8f wmma_bf16(v16bf a, v16bf b, v8f c) {
  // D = A(16x32 bf16) x B(32x16 bf16) + C(16x16 f32)
  return __builtin_amdgcn_wmma_f32_16x16x32_bf16(false, a, false, b, (short)0, c,
                                                 false, false);
}

// ---------------------------------------------------------------------------
// Generic tiled GEMM: C[M,N] = A[M,K] @ B[K,N]  (row-major f32 in, bf16 WMMA,
// f32 accumulate). 128x128 tile, BK=32, 256 threads = 8 waves (2x4 wave grid,
// each wave owns 64x32). Software-pipelined: next K-step's globals are fetched
// into registers while WMMAs consume the current LDS tiles. Optional second
// accumulation segment (A2@B2) for the fused dense+tanh epilogue.
// EPI: 0 = store f32, 1 = store exp(f32), 2 = tanh(acc + b0[n] + b1[n]) -> bf16
// ---------------------------------------------------------------------------
template<int EPI>
__global__ void __launch_bounds__(256) gemm_bf16_kernel(
    const float* __restrict__ A, long sA, int aDiv,
    const float* __restrict__ Bm, long sB,
    const float* __restrict__ A2, long sA2, const float* __restrict__ B2, int K2,
    int M, int N, int K,
    float* __restrict__ C, __bf16* __restrict__ Cb, long sC,
    const float* __restrict__ bias0, const float* __restrict__ bias1)
{
  __shared__ __bf16 As[128 * 32];
  __shared__ __bf16 Bs[128 * 32];   // stored transposed: Bs[n][k]

  const int tid   = threadIdx.x;
  const int bz    = blockIdx.z;
  const int m0    = blockIdx.y * 128;
  const int n0    = blockIdx.x * 128;
  const int wave  = tid >> 5;
  const int lane  = tid & 31;
  const int l15   = lane & 15;
  const int khalf = lane >> 4;
  const int wm    = wave >> 2;   // 0..1 -> 64 rows each
  const int wn    = wave & 3;    // 0..3 -> 32 cols each

  // per-thread staging coordinates
  const int arow  = tid >> 1;
  const int acol0 = (tid & 1) * 16;
  const int bkin  = tid >> 3;
  const int bcol0 = (tid & 7) * 16;

  v8f acc[4][2];
  #pragma unroll
  for (int at = 0; at < 4; ++at)
    #pragma unroll
    for (int bt = 0; bt < 2; ++bt)
      #pragma unroll
      for (int v = 0; v < 8; ++v) acc[at][bt][v] = 0.0f;

  // segment list (flattened two-phase accumulation)
  const float* segA[2]; const float* segB[2]; int segK[2];
  segA[0] = A + (long)(bz / aDiv) * sA;
  segB[0] = Bm + (long)bz * sB;
  segK[0] = K;
  int nseg = 1;
  if (A2 != nullptr) {
    segA[1] = A2 + (long)bz * sA2;
    segB[1] = B2;
    segK[1] = K2;
    nseg = 2;
  }

  auto load_tiles = [&](int s, int k0, float* ta, float* tb) {
    const float* Ap = segA[s]; const float* Bp = segB[s]; int Kc = segK[s];
    int m = m0 + arow;
    if (m < M) {
      const float* p = Ap + (long)m * Kc + k0 + acol0;
      #pragma unroll
      for (int u = 0; u < 16; ++u) ta[u] = p[u];
      if (k0 + 32 < Kc) __builtin_prefetch(p + 32, 0, 1);
    } else {
      #pragma unroll
      for (int u = 0; u < 16; ++u) ta[u] = 0.0f;
    }
    const float* pb = Bp + (long)(k0 + bkin) * N + n0 + bcol0;
    #pragma unroll
    for (int u = 0; u < 16; ++u) {
      int n = n0 + bcol0 + u;
      tb[u] = (n < N) ? pb[u] : 0.0f;
    }
    if (k0 + 32 < Kc) __builtin_prefetch(pb + 32L * N, 0, 1);
  };

  float ta[16], tb[16];
  int s = 0, k0 = 0;
  load_tiles(0, 0, ta, tb);

  while (true) {
    { // commit staged tiles to LDS (f32 -> bf16)
      __bf16* da = &As[arow * 32 + acol0];
      #pragma unroll
      for (int u = 0; u < 16; ++u) da[u] = (__bf16)ta[u];
      #pragma unroll
      for (int u = 0; u < 16; ++u) Bs[(bcol0 + u) * 32 + bkin] = (__bf16)tb[u];
    }
    __syncthreads();

    // advance iterator; kick off next tile's global loads before the WMMAs
    int ns = s, nk = k0 + 32;
    if (nk >= segK[s]) { ns = s + 1; nk = 0; }
    const bool more = (ns < nseg);
    float ta2[16], tb2[16];
    if (more) load_tiles(ns, nk, ta2, tb2);

    // A fragment: lane l15 = row m; frag[0..7] = K khalf*8+0..7,
    // frag[8..15] = K 16+khalf*8+0..7
    v16bf afr[4];
    #pragma unroll
    for (int at = 0; at < 4; ++at) {
      const __bf16* pr = &As[(wm * 64 + at * 16 + l15) * 32 + khalf * 8];
      #pragma unroll
      for (int t = 0; t < 8; ++t) {
        afr[at][t]     = pr[t];
        afr[at][8 + t] = pr[16 + t];
      }
    }
    // B fragment: lane l15 = col n; frag[0..7] = K khalf*8+0..7,
    // frag[8..15] = K 16+khalf*8+0..7
    v16bf bfr[2];
    #pragma unroll
    for (int bt = 0; bt < 2; ++bt) {
      const __bf16* pr = &Bs[(wn * 32 + bt * 16 + l15) * 32 + khalf * 8];
      #pragma unroll
      for (int t = 0; t < 8; ++t) { bfr[bt][t] = pr[t]; bfr[bt][8 + t] = pr[16 + t]; }
    }
    #pragma unroll
    for (int at = 0; at < 4; ++at)
      #pragma unroll
      for (int bt = 0; bt < 2; ++bt)
        acc[at][bt] = wmma_bf16(afr[at], bfr[bt], acc[at][bt]);
    __syncthreads();

    if (!more) break;
    #pragma unroll
    for (int u = 0; u < 16; ++u) { ta[u] = ta2[u]; tb[u] = tb2[u]; }
    s = ns; k0 = nk;
  }

  // Epilogue. C/D layout: lane l15 = n; VGPR v: m = khalf*8 + v.
  #pragma unroll
  for (int at = 0; at < 4; ++at) {
    #pragma unroll
    for (int bt = 0; bt < 2; ++bt) {
      int n = n0 + wn * 32 + bt * 16 + l15;
      #pragma unroll
      for (int v = 0; v < 8; ++v) {
        int m = m0 + wm * 64 + at * 16 + khalf * 8 + v;
        if (m < M && n < N) {
          float val = acc[at][bt][v];
          if (EPI == 1) val = expf(val);
          if (EPI == 2) {
            val = tanhf(val + bias0[n] + bias1[n]);
            Cb[(long)bz * sC + (long)m * N + n] = (__bf16)val;
          } else {
            C[(long)bz * sC + (long)m * N + n] = val;
          }
        }
      }
    }
  }
}

// ---------------------------------------------------------------------------
// Group bilinear classifier via WMMA, g-split partials.
// For a K=32 feature chunk (g,i,j0..j0+31) the A fragment is rank-1:
// A[p,k] = h[p,g,i] * t[p,g,j0+k] (built in registers, v_pk_mul_bf16).
// The (g,i) weight slice (112 r x 64 j bf16, rows padded to 72 to avoid LDS
// bank conflicts, r>=97 zero-filled) is staged once in LDS and shared by all
// 8 waves; each wave owns 2 pair-tiles (32 pairs) x 7 r-tiles.
// blockIdx.y in {0,1} selects 6 of the 12 groups; partial f32 sums go to ws.
// ---------------------------------------------------------------------------
__global__ void __launch_bounds__(256) bilinear_partial_kernel(
    const __bf16* __restrict__ Hh, const __bf16* __restrict__ Th,
    const __bf16* __restrict__ Wb, float* __restrict__ part)
{
  __shared__ __bf16 Ws[RPAD * WSTRIDE];   // 112 x 72 bf16 = 16128 B

  const int tid   = threadIdx.x;
  const int wave  = tid >> 5;
  const int lane  = tid & 31;
  const int l15   = lane & 15;
  const int khalf = lane >> 4;
  const int base  = blockIdx.x * 256;           // 256 pairs per block
  const int g0    = blockIdx.y * (GG / 2);      // 6 groups per block
  const int p0a   = base + (wave * 2 + 0) * 16;
  const int p0b   = base + (wave * 2 + 1) * 16;
  const int paa   = p0a + l15;
  const int pab   = p0b + l15;

  v8f acc[2][7];
  #pragma unroll
  for (int pt = 0; pt < 2; ++pt)
    #pragma unroll
    for (int rt = 0; rt < 7; ++rt)
      #pragma unroll
      for (int v = 0; v < 8; ++v) acc[pt][rt][v] = 0.0f;

  const __bf16* hrow[2] = { Hh + (long)paa * HH, Hh + (long)pab * HH };
  const __bf16* trow[2] = { Th + (long)paa * HH, Th + (long)pab * HH };

  for (int g = g0; g < g0 + GG / 2; ++g) {
    for (int i = 0; i < BSs; ++i) {
      // ---- stage W[r, g,i, j] slice into LDS (112 x 64, zero-pad r>=97) ----
      #pragma unroll
      for (int u = 0; u < 4; ++u) {
        int c = tid + 256 * u;                    // 896 chunks of 8 bf16
        if (c < (RPAD * BSs) / 8) {
          int r  = c >> 3;
          int j0 = (c & 7) * 8;
          __bf16* dst = &Ws[r * WSTRIDE + j0];
          if (r < RR) {
            const __bf16* src = Wb + (long)r * (HH * BSs) + g * 4096 + i * 64 + j0;
            #pragma unroll
            for (int t = 0; t < 8; ++t) dst[t] = src[t];
            if (i + 1 < BSs) __builtin_prefetch(src + 64, 0, 1);
          } else {
            #pragma unroll
            for (int t = 0; t < 8; ++t) dst[t] = (__bf16)0.0f;
          }
        }
      }
      __syncthreads();

      // ---- compute ----
      float hs[2] = { (float)hrow[0][g * 64 + i], (float)hrow[1][g * 64 + i] };
      #pragma unroll
      for (int j0 = 0; j0 < 64; j0 += 32) {
        v16bf afr[2];
        #pragma unroll
        for (int pt = 0; pt < 2; ++pt) {
          const __bf16* tp = trow[pt] + g * 64 + j0 + khalf * 8;
          #pragma unroll
          for (int t = 0; t < 8; ++t) {
            afr[pt][t]     = (__bf16)(hs[pt] * (float)tp[t]);
            afr[pt][8 + t] = (__bf16)(hs[pt] * (float)tp[16 + t]);
          }
        }
        #pragma unroll
        for (int rt = 0; rt < 7; ++rt) {
          const __bf16* pr = &Ws[(rt * 16 + l15) * WSTRIDE + j0 + khalf * 8];
          v16bf bfr;
          #pragma unroll
          for (int t = 0; t < 8; ++t) { bfr[t] = pr[t]; bfr[8 + t] = pr[16 + t]; }
          #pragma unroll
          for (int pt = 0; pt < 2; ++pt)
            acc[pt][rt] = wmma_bf16(afr[pt], bfr, acc[pt][rt]);
        }
      }
      __syncthreads();
    }
  }

  // partial store (f32, no bias/mask yet). C/D: lane l15 = r; m = khalf*8+v.
  float* pb = part + (long)blockIdx.y * BPALL * RR;
  #pragma unroll
  for (int pt = 0; pt < 2; ++pt) {
    const int pbase = (pt == 0) ? p0a : p0b;
    #pragma unroll
    for (int rt = 0; rt < 7; ++rt) {
      int r = rt * 16 + l15;
      if (r >= RR) continue;
      #pragma unroll
      for (int v = 0; v < 8; ++v) {
        int p = pbase + khalf * 8 + v;
        pb[(long)p * RR + r] = acc[pt][rt][v];
      }
    }
  }
}

__global__ void bilinear_combine_kernel(const float* __restrict__ part,
                                        const float* __restrict__ cb,
                                        const unsigned char* __restrict__ tmask,
                                        float* __restrict__ out) {
  long idx = (long)blockIdx.x * 256 + threadIdx.x;
  if (idx >= (long)BPALL * RR) return;
  int r = (int)(idx % RR);
  float val = part[idx] + part[idx + (long)BPALL * RR] + cb[r];
  float msk = tmask[idx] ? 1.0f : 0.0f;
  out[idx] = val - (1.0f - msk) * MAXV;
}

// ------------------------- small glue kernels ------------------------------
__global__ void transpose4_kernel(const float* __restrict__ w0,
                                  const float* __restrict__ w1,
                                  const float* __restrict__ w2,
                                  const float* __restrict__ w3,
                                  float* __restrict__ out) {
  long idx = (long)blockIdx.x * 256 + threadIdx.x;
  if (idx >= 4L * HH * HH) return;
  int w   = (int)(idx / (HH * HH));
  int rem = (int)(idx % (HH * HH));
  int k = rem / HH, n = rem % HH;
  const float* src = (w == 0) ? w0 : (w == 1) ? w1 : (w == 2) ? w2 : w3;
  out[(long)w * HH * HH + (long)k * HH + n] = src[(long)n * HH + k];
}

__global__ void cvt_clas_kernel(const float* __restrict__ w, __bf16* __restrict__ o) {
  long i = (long)blockIdx.x * 256 + threadIdx.x;
  if (i < (long)RR * HH * BSs) o[i] = (__bf16)w[i];
}

__global__ void entity_log_kernel(const float* __restrict__ emap, float* __restrict__ ent) {
  __shared__ float red[256];
  int row = blockIdx.x;         // b*EE + e
  int tid = threadIdx.x;
  float s = (tid < MM) ? emap[(long)row * MM + tid] : 0.0f;
  red[tid] = s; __syncthreads();
  for (int o = 128; o > 0; o >>= 1) {
    if (tid < o) red[tid] += red[tid + o];
    __syncthreads();
  }
  bool zero = (red[0] == 0.0f);
  for (int h = tid; h < HH; h += 256) {
    float v = ent[(long)row * HH + h];
    ent[(long)row * HH + h] = logf(zero ? 1.0f : v);
  }
}

__global__ void em_norm_kernel(float* __restrict__ em) {
  __shared__ float red[256];
  int row = blockIdx.x;         // b*EE + e
  int tid = threadIdx.x;
  float vals[4]; float s = 0.0f;
  #pragma unroll
  for (int c = 0; c < 4; ++c) { vals[c] = em[(long)row * LL + c * 256 + tid]; s += vals[c]; }
  red[tid] = s; __syncthreads();
  for (int o = 128; o > 0; o >>= 1) {
    if (tid < o) red[tid] += red[tid + o];
    __syncthreads();
  }
  float inv = 1.0f / (red[0] + MINV);
  #pragma unroll
  for (int c = 0; c < 4; ++c) em[(long)row * LL + c * 256 + tid] = vals[c] * inv;
}

__global__ void ca_kernel(const float* __restrict__ ea, const int* __restrict__ hts,
                          float* __restrict__ ca) {
  __shared__ float red[256];
  int bp = blockIdx.x;          // b*PP + p
  int tid = threadIdx.x;
  int b  = bp >> 11;            // PP = 2048
  int hi = hts[bp * 2], ti = hts[bp * 2 + 1];
  float mask = (hi + ti != 0) ? 1.0f : 0.0f;   // folds ht_mask into ca rows
  const float* eab = ea + (long)b * NHEADS * EE * LL;
  float vals[4]; float s = 0.0f;
  #pragma unroll
  for (int c = 0; c < 4; ++c) {
    int l = c * 256 + tid; float t = 0.0f;
    #pragma unroll
    for (int hd = 0; hd < NHEADS; ++hd)
      t += eab[((long)hd * EE + hi) * LL + l] * eab[((long)hd * EE + ti) * LL + l];
    vals[c] = t; s += t;
  }
  red[tid] = s; __syncthreads();
  for (int o = 128; o > 0; o >>= 1) {
    if (tid < o) red[tid] += red[tid + o];
    __syncthreads();
  }
  float inv = mask / (red[0] + MINV);
  #pragma unroll
  for (int c = 0; c < 4; ++c) ca[(long)bp * LL + c * 256 + tid] = vals[c] * inv;
}

__global__ void gather_ht_kernel(const float* __restrict__ ent, const int* __restrict__ hts,
                                 float* __restrict__ hg, float* __restrict__ tg) {
  int bp = blockIdx.x;
  int tid = threadIdx.x;
  int b  = bp >> 11;
  int hi = hts[bp * 2], ti = hts[bp * 2 + 1];
  float mask = (hi + ti != 0) ? 1.0f : 0.0f;
  const float* hrow = ent + ((long)b * EE + hi) * HH;
  const float* trow = ent + ((long)b * EE + ti) * HH;
  #pragma unroll
  for (int c = 0; c < 3; ++c) {                 // HH = 3*256
    int h = c * 256 + tid;
    hg[(long)bp * HH + h] = mask * hrow[h];
    tg[(long)bp * HH + h] = mask * trow[h];
  }
}

// ---------------------------------------------------------------------------
extern "C" void kernel_launch(void* const* d_in, const int* in_sizes, int n_in,
                              void* d_out, int out_size, void* d_ws, size_t ws_size,
                              hipStream_t stream) {
  (void)in_sizes; (void)n_in; (void)out_size; (void)ws_size;
  const float* context   = (const float*)d_in[0];
  const float* attention = (const float*)d_in[1];
  const float* mmap      = (const float*)d_in[2];
  const float* emap      = (const float*)d_in[3];
  const float* h_w  = (const float*)d_in[4];  const float* h_b  = (const float*)d_in[5];
  const float* t_w  = (const float*)d_in[6];  const float* t_b  = (const float*)d_in[7];
  const float* hc_w = (const float*)d_in[8];  const float* hc_b = (const float*)d_in[9];
  const float* tc_w = (const float*)d_in[10]; const float* tc_b = (const float*)d_in[11];
  const float* clas_w = (const float*)d_in[12];
  const float* clas_b = (const float*)d_in[13];
  const int*   hts    = (const int*)d_in[14];
  const unsigned char* tmask = (const unsigned char*)d_in[15];
  float* out = (float*)d_out;

  char* ws = (char*)d_ws;
  size_t off = 0;
  auto alloc = [&](size_t bytes) -> void* {
    void* p = ws + off;
    off = (off + bytes + 255) & ~(size_t)255;
    return p;
  };

  float*  mention = (float*)alloc((size_t)Bb * MM * HH * 4);
  float*  entity  = (float*)alloc((size_t)Bb * EE * HH * 4);
  float*  em      = (float*)alloc((size_t)Bb * EE * LL * 4);
  float*  ea      = (float*)alloc((size_t)Bb * NHEADS * EE * LL * 4);
  float*  ca      = (float*)alloc((size_t)Bb * PP * LL * 4);
  float*  ci      = (float*)alloc((size_t)Bb * PP * HH * 4);
  float*  hg      = (float*)alloc((size_t)Bb * PP * HH * 4);
  float*  tg      = (float*)alloc((size_t)Bb * PP * HH * 4);
  float*  wT      = (float*)alloc((size_t)4 * HH * HH * 4);
  __bf16* clasbf  = (__bf16*)alloc((size_t)RR * HH * BSs * 2);
  __bf16* hhid    = (__bf16*)alloc((size_t)Bb * PP * HH * 2);
  __bf16* thid    = (__bf16*)alloc((size_t)Bb * PP * HH * 2);
  float*  blpart  = (float*)alloc((size_t)2 * BPALL * RR * 4);

  // weight prep
  {
    long n = 4L * HH * HH;
    transpose4_kernel<<<(unsigned)((n + 255) / 256), 256, 0, stream>>>(h_w, t_w, hc_w, tc_w, wT);
  }
  {
    long n = (long)RR * HH * BSs;
    cvt_clas_kernel<<<(unsigned)((n + 255) / 256), 256, 0, stream>>>(clas_w, clasbf);
  }

  // mention = exp(mmap @ context)            [B,128,768]
  gemm_bf16_kernel<1><<<dim3(HH / 128, 1, Bb), 256, 0, stream>>>(
      mmap, (long)MM * LL, 1, context, (long)LL * HH,
      nullptr, 0, nullptr, 0,
      MM, HH, LL, mention, nullptr, (long)MM * HH, nullptr, nullptr);

  // entity = emap @ mention                  [B,64,768]
  gemm_bf16_kernel<0><<<dim3(HH / 128, 1, Bb), 256, 0, stream>>>(
      emap, (long)EE * MM, 1, mention, (long)MM * HH,
      nullptr, 0, nullptr, 0,
      EE, HH, MM, entity, nullptr, (long)EE * HH, nullptr, nullptr);

  entity_log_kernel<<<Bb * EE, 256, 0, stream>>>(emap, entity);

  // em = emap @ mmap                         [B,64,1024]
  gemm_bf16_kernel<0><<<dim3(LL / 128, 1, Bb), 256, 0, stream>>>(
      emap, (long)EE * MM, 1, mmap, (long)MM * LL,
      nullptr, 0, nullptr, 0,
      EE, LL, MM, em, nullptr, (long)EE * LL, nullptr, nullptr);

  em_norm_kernel<<<Bb * EE, 256, 0, stream>>>(em);

  // ea = em @ attention   (48 batches; A shared across the 12 heads of a b)
  gemm_bf16_kernel<0><<<dim3(LL / 128, 1, Bb * NHEADS), 256, 0, stream>>>(
      em, (long)EE * LL, NHEADS, attention, (long)LL * LL,
      nullptr, 0, nullptr, 0,
      EE, LL, LL, ea, nullptr, (long)EE * LL, nullptr, nullptr);

  ca_kernel<<<Bb * PP, 256, 0, stream>>>(ea, hts, ca);

  // ci = ca @ context                        [B,2048,768] (ht_mask folded in ca)
  gemm_bf16_kernel<0><<<dim3(HH / 128, PP / 128, Bb), 256, 0, stream>>>(
      ca, (long)PP * LL, 1, context, (long)LL * HH,
      nullptr, 0, nullptr, 0,
      PP, HH, LL, ci, nullptr, (long)PP * HH, nullptr, nullptr);

  gather_ht_kernel<<<Bb * PP, 256, 0, stream>>>(entity, hts, hg, tg);

  const float* hwT = wT + 0L * HH * HH;
  const float* twT = wT + 1L * HH * HH;
  const float* hcT = wT + 2L * HH * HH;
  const float* tcT = wT + 3L * HH * HH;

  // hhid = tanh(hg@h_w^T + ci@hc_w^T + h_b + hc_b)  -> bf16
  gemm_bf16_kernel<2><<<dim3(HH / 128, PP / 128, Bb), 256, 0, stream>>>(
      hg, (long)PP * HH, 1, hwT, 0L,
      ci, (long)PP * HH, hcT, HH,
      PP, HH, HH, nullptr, hhid, (long)PP * HH, h_b, hc_b);

  // thid = tanh(tg@t_w^T + ci@tc_w^T + t_b + tc_b)  -> bf16
  gemm_bf16_kernel<2><<<dim3(HH / 128, PP / 128, Bb), 256, 0, stream>>>(
      tg, (long)PP * HH, 1, twT, 0L,
      ci, (long)PP * HH, tcT, HH,
      PP, HH, HH, nullptr, thid, (long)PP * HH, t_b, tc_b);

  // group bilinear: g-split partials, then combine with bias + type-mask
  bilinear_partial_kernel<<<dim3(BPALL / 256, 2), 256, 0, stream>>>(hhid, thid,
                                                                    clasbf, blpart);
  {
    long n = (long)BPALL * RR;
    bilinear_combine_kernel<<<(unsigned)((n + 255) / 256), 256, 0, stream>>>(
        blpart, clas_b, tmask, out);
  }
}
